// LlamaAttentionWrapperQK_84997402788411
// MI455X (gfx1250) — matrix operand order repaired and checked by reading, not compiled, not run on hardware
//
#include <hip/hip_runtime.h>
#include <hip/hip_bf16.h>

// LlamaAttention forward, B=1 S=2048 H=32 D=128, gfx1250 (MI455X).
//   K0 convert_rows  : fp32 -> bf16 (permute_W folded into Wq/Wk)
//   K1 gemm128<false>: QKV projections, bf16 WMMA + async-LDS double buffer,
//                      interleaved RoPE epilogue (DPP partner exchange)
//   K2 flash_attn    : online-softmax flash attention, WMMA + DPP reductions
//   K3 gemm128<true> : output projection -> fp32
// Workspace: [hbf 16M][wq 32M][wk 32M][wv 32M][wo 32M][q 16M][k 16M][v 16M][attn 16M]

typedef __bf16 bf16;
typedef __attribute__((ext_vector_type(8)))  __bf16 v8bf;
typedef __attribute__((ext_vector_type(16))) __bf16 v16bf;
typedef __attribute__((ext_vector_type(8)))  float  v8f;

#define S_LEN 2048
#define NHEAD 32
#define DHEAD 128
#define HID   4096

// ---------------------------------------------------------------------------
// helpers
// ---------------------------------------------------------------------------
__device__ __forceinline__ unsigned short f2bf_bits(float f) {
  unsigned int u = __builtin_bit_cast(unsigned int, f);
  u += 0x7FFFu + ((u >> 16) & 1u);  // RNE
  return (unsigned short)(u >> 16);
}
__device__ __forceinline__ bf16 f2bf(float f) {
  unsigned short b = f2bf_bits(f);
  return __builtin_bit_cast(bf16, b);
}

// DPP16 ROW_XMASK lane exchange (lane i <- lane i^mask within 16-lane row),
// pure VALU: avoids ds_bpermute + s_wait_dscnt round trips for reductions.
template <int CTRL>
__device__ __forceinline__ float dpp_xor(float x) {
#if __has_builtin(__builtin_amdgcn_update_dpp)
  return __builtin_bit_cast(
      float, __builtin_amdgcn_update_dpp(0, __builtin_bit_cast(int, x), CTRL,
                                         0xf, 0xf, false));
#else
  return __shfl_xor(x, CTRL & 0xF);
#endif
}
__device__ __forceinline__ float red_max16(float x) {
  x = fmaxf(x, dpp_xor<0x161>(x));   // xor 1
  x = fmaxf(x, dpp_xor<0x162>(x));   // xor 2
  x = fmaxf(x, dpp_xor<0x164>(x));   // xor 4
  x = fmaxf(x, dpp_xor<0x168>(x));   // xor 8
  return x;
}
__device__ __forceinline__ float red_sum16(float x) {
  x += dpp_xor<0x161>(x);
  x += dpp_xor<0x162>(x);
  x += dpp_xor<0x164>(x);
  x += dpp_xor<0x168>(x);
  return x;
}

// ---------------------------------------------------------------------------
// async global -> LDS (ASYNCcnt path), guarded so either toolchain compiles.
// Builtin expects int4 pointers: global (AS1) source, LDS (AS3) destination.
// ---------------------------------------------------------------------------
#if __has_builtin(__builtin_amdgcn_global_load_async_to_lds_b128)
#define HAVE_ASYNC_LDS 1
typedef __attribute__((ext_vector_type(4))) int int4v;
typedef __attribute__((address_space(1))) int4v GAS4;
typedef __attribute__((address_space(3))) int4v LAS4;
__device__ __forceinline__ void async_cp16(const bf16* g, bf16* l) {
  // generic->AS1 via ptrtoint/inttoptr; LDS generic addr low 32 bits = offset
  __builtin_amdgcn_global_load_async_to_lds_b128(
      (GAS4*)(unsigned long long)g, (LAS4*)(unsigned)(unsigned long long)l, 0, 0);
}
__device__ __forceinline__ void wait_async0() {
#if __has_builtin(__builtin_amdgcn_s_wait_asynccnt)
  __builtin_amdgcn_s_wait_asynccnt(0);
#else
  asm volatile("s_wait_asynccnt 0x0" ::: "memory");
#endif
}
#else
#define HAVE_ASYNC_LDS 0
#endif

// ---------------------------------------------------------------------------
// WMMA fragment loads (layouts per CDNA5 ISA §7.12.2)
// ---------------------------------------------------------------------------
__device__ __forceinline__ v16bf loadA(const bf16* rowp, int hi) {
  v8bf a = *reinterpret_cast<const v8bf*>(rowp + hi * 8);
  v8bf b = *reinterpret_cast<const v8bf*>(rowp + 16 + hi * 8);
  return __builtin_shufflevector(a, b, 0,1,2,3,4,5,6,7,8,9,10,11,12,13,14,15);
}
__device__ __forceinline__ v16bf loadB(const bf16* rowp, int hi) {
  v8bf a = *reinterpret_cast<const v8bf*>(rowp + hi * 8);
  v8bf b = *reinterpret_cast<const v8bf*>(rowp + 16 + hi * 8);
  return __builtin_shufflevector(a, b, 0,1,2,3,4,5,6,7,8,9,10,11,12,13,14,15);
}
__device__ __forceinline__ v8f wmma_bf16(v16bf a, v16bf b, v8f c) {
  return __builtin_amdgcn_wmma_f32_16x16x32_bf16(false, a, false, b, (short)0, c,
                                                 false, false);
}

// ---------------------------------------------------------------------------
// K0: fp32 -> bf16 rows; permute==1 folds Llama permute_W
// ---------------------------------------------------------------------------
__global__ __launch_bounds__(256)
void convert_rows(const float* __restrict__ src, bf16* __restrict__ dst, int permute) {
  const int row = blockIdx.x;
  int srow = row;
  if (permute) {
    const int h = row >> 7, d = row & 127;
    const int sd = (d & 1) ? ((d >> 1) + 64) : (d >> 1);
    srow = (h << 7) | sd;
  }
  const float4* sp = reinterpret_cast<const float4*>(src + (size_t)srow * HID);
  bf16* dp = dst + (size_t)row * HID;
  for (int i = threadIdx.x; i < HID / 4; i += 256) {
    float4 v = sp[i];
    unsigned long long pk =
        (unsigned long long)f2bf_bits(v.x) |
        ((unsigned long long)f2bf_bits(v.y) << 16) |
        ((unsigned long long)f2bf_bits(v.z) << 32) |
        ((unsigned long long)f2bf_bits(v.w) << 48);
    *reinterpret_cast<unsigned long long*>(dp + i * 4) = pk;
  }
}

// ---------------------------------------------------------------------------
// K1/K3: 128x128 block GEMM, BK=64 slabs, double-buffered async LDS pipeline.
// 8 waves: 4(M) x 2(N), wave tile 32x64 -> 8 fp32 accumulators; 16 WMMA/slab.
// ---------------------------------------------------------------------------
#define NSLAB (HID / 64)

template <bool FINAL>
__global__ __launch_bounds__(256)
void gemm128(const bf16* __restrict__ X,
             const bf16* __restrict__ W0, const bf16* __restrict__ W1,
             const bf16* __restrict__ W2,
             bf16* __restrict__ O0, bf16* __restrict__ O1, bf16* __restrict__ O2,
             float* __restrict__ OF) {
  __shared__ bf16 As[2][128][72];  // [buf][m][k], pad 8 keeps 16B alignment
  __shared__ bf16 Bs[2][128][72];  // [buf][n][k]

  const int z = FINAL ? 0 : blockIdx.z;
  const bf16* W = FINAL ? W0 : (z == 0 ? W0 : (z == 1 ? W1 : W2));
  const int gm0 = blockIdx.y * 128;
  const int gn0 = blockIdx.x * 128;
  const int tid = threadIdx.x;
  const int lane = tid & 31, wid = tid >> 5;
  const int r = lane & 15, hi = lane >> 4;
  const int wm = wid & 3, wn = wid >> 2;
  const int lrow = tid >> 1;          // 0..127
  const int lseg = (tid & 1) * 32;    // 0 or 32

  const v8f z8 = {0.f, 0.f, 0.f, 0.f, 0.f, 0.f, 0.f, 0.f};
  v8f acc[2][4];
#pragma unroll
  for (int mt = 0; mt < 2; ++mt)
#pragma unroll
    for (int nt = 0; nt < 4; ++nt) acc[mt][nt] = z8;

  const bf16* Xrow = X + (size_t)(gm0 + lrow) * HID + lseg;
  const bf16* Wrow = W + (size_t)(gn0 + lrow) * HID + lseg;

#if HAVE_ASYNC_LDS
#pragma unroll
  for (int j = 0; j < 4; ++j) {  // prologue: slab 0 in flight
    async_cp16(Xrow + j * 8, &As[0][lrow][lseg + j * 8]);
    async_cp16(Wrow + j * 8, &Bs[0][lrow][lseg + j * 8]);
  }
#endif

  for (int s = 0; s < NSLAB; ++s) {
    const int cur = s & 1;
#if HAVE_ASYNC_LDS
    wait_async0();        // my slab-s deposits complete
    __syncthreads();      // everyone's deposits visible; prev slab fully read
    if (s + 1 < NSLAB) {  // overlap slab s+1 copy with slab s math
      const int k1 = (s + 1) * 64;
#pragma unroll
      for (int j = 0; j < 4; ++j) {
        async_cp16(Xrow + k1 + j * 8, &As[cur ^ 1][lrow][lseg + j * 8]);
        async_cp16(Wrow + k1 + j * 8, &Bs[cur ^ 1][lrow][lseg + j * 8]);
      }
    }
#else
    const int k0 = s * 64;
    uint4 a[4], b[4];
#pragma unroll
    for (int j = 0; j < 4; ++j) {
      a[j] = *reinterpret_cast<const uint4*>(Xrow + k0 + j * 8);
      b[j] = *reinterpret_cast<const uint4*>(Wrow + k0 + j * 8);
    }
    __syncthreads();
#pragma unroll
    for (int j = 0; j < 4; ++j) {
      *reinterpret_cast<uint4*>(&As[cur][lrow][lseg + j * 8]) = a[j];
      *reinterpret_cast<uint4*>(&Bs[cur][lrow][lseg + j * 8]) = b[j];
    }
    __syncthreads();
#endif

#pragma unroll
    for (int ks = 0; ks < 2; ++ks) {
      v16bf af[2], bfv[4];
#pragma unroll
      for (int mt = 0; mt < 2; ++mt)
        af[mt] = loadA(&As[cur][wm * 32 + mt * 16 + r][ks * 32], hi);
#pragma unroll
      for (int nt = 0; nt < 4; ++nt)
        bfv[nt] = loadB(&Bs[cur][wn * 64 + nt * 16 + r][ks * 32], hi);
#pragma unroll
      for (int mt = 0; mt < 2; ++mt)
#pragma unroll
        for (int nt = 0; nt < 4; ++nt)
          acc[mt][nt] = wmma_bf16(af[mt], bfv[nt], acc[mt][nt]);
    }
  }

  if (FINAL) {
#pragma unroll
    for (int mt = 0; mt < 2; ++mt)
#pragma unroll
      for (int nt = 0; nt < 4; ++nt) {
        const int n = gn0 + wn * 64 + nt * 16 + r;
#pragma unroll
        for (int i = 0; i < 8; ++i) {
          const int s = gm0 + wm * 32 + mt * 16 + hi * 8 + i;
          OF[(size_t)s * HID + n] = acc[mt][nt][i];
        }
      }
  } else {
    bf16* Out = (z == 0) ? O0 : ((z == 1) ? O1 : O2);
    const bool rope = (z < 2);
#pragma unroll
    for (int mt = 0; mt < 2; ++mt)
#pragma unroll
      for (int nt = 0; nt < 4; ++nt) {
        const int n = gn0 + wn * 64 + nt * 16 + r;
        const int h = n >> 7, d = n & 127;
        // inv_freq = 10000^(-2(d>>1)/128) = 2^(-(d>>1)*log2(10000)/64)
        const float invf = exp2f((float)(d >> 1) * -0.2076205059304601f);
#pragma unroll
        for (int i = 0; i < 8; ++i) {
          const int s = gm0 + wm * 32 + mt * 16 + hi * 8 + i;
          float v = acc[mt][nt][i];
          float pv = dpp_xor<0x161>(v);  // partner dim d^1 in adjacent lane
          float o = v;
          if (rope) {
            float c, sn;
            __sincosf((float)s * invf, &sn, &c);
            o = (d & 1) ? fmaf(v, c, pv * sn)    // odd:  x_o*c + x_e*s
                        : fmaf(v, c, -pv * sn);  // even: x_e*c - x_o*s
          }
          Out[((size_t)h * S_LEN + s) * DHEAD + d] = f2bf(o);
        }
      }
  }
}

// ---------------------------------------------------------------------------
// K2: flash attention. CTA = (128 q rows, head); 8 waves own 16 q rows each;
// 32-key slabs; online softmax with DPP row reductions; all matmuls WMMA.
// ---------------------------------------------------------------------------
__global__ __launch_bounds__(256)
void flash_attn(const bf16* __restrict__ Q, const bf16* __restrict__ K,
                const bf16* __restrict__ V, bf16* __restrict__ Oa) {
  __shared__ bf16 Qs[128][136];   // [q][d]
  __shared__ bf16 Ks[32][136];    // [key][d]
  __shared__ bf16 Vt[128][40];    // [d][key] transposed V tile
  __shared__ bf16 Ps[8][16][40];  // per-wave P relayout scratch

  const int qb = blockIdx.x, h = blockIdx.y;
  const int tid = threadIdx.x;
  const int lane = tid & 31, wid = tid >> 5;
  const int r = lane & 15, hi = lane >> 4;

  {  // Q block 128x128 -> LDS
    const int row = tid >> 1;
    const bf16* qrow = Q + ((size_t)h * S_LEN + qb * 128 + row) * DHEAD;
#pragma unroll
    for (int p = 0; p < 4; ++p) {
      const int seg = (tid & 1) * 16 + p * 32;
#if HAVE_ASYNC_LDS
      async_cp16(qrow + seg, &Qs[row][seg]);
      async_cp16(qrow + seg + 8, &Qs[row][seg + 8]);
#else
      uint4 q0 = *reinterpret_cast<const uint4*>(qrow + seg);
      uint4 q1 = *reinterpret_cast<const uint4*>(qrow + seg + 8);
      *reinterpret_cast<uint4*>(&Qs[row][seg])     = q0;
      *reinterpret_cast<uint4*>(&Qs[row][seg + 8]) = q1;
#endif
    }
  }

  const v8f z8 = {0.f, 0.f, 0.f, 0.f, 0.f, 0.f, 0.f, 0.f};
  v8f oacc[8];
#pragma unroll
  for (int dt = 0; dt < 8; ++dt) oacc[dt] = z8;
  float mrow[8], lrow[8];
#pragma unroll
  for (int i = 0; i < 8; ++i) { mrow[i] = -3.0e38f; lrow[i] = 0.f; }

  const float scale = 0.08838834764831845f;  // 1/sqrt(128)
  const int nkb = qb * 4 + 4;                // causal bound
  const int keyld = tid >> 3;                // 0..31
  const int segld = (tid & 7) * 16;          // 0..112

  for (int kb = 0; kb < nkb; ++kb) {
    const bf16* krow = K + ((size_t)h * S_LEN + kb * 32 + keyld) * DHEAD + segld;
    const bf16* vrow = V + ((size_t)h * S_LEN + kb * 32 + keyld) * DHEAD + segld;
    union { uint4 u[2]; bf16 b[16]; } vv;
    vv.u[0] = *reinterpret_cast<const uint4*>(vrow);
    vv.u[1] = *reinterpret_cast<const uint4*>(vrow + 8);
    __syncthreads();  // previous slab fully consumed
#if HAVE_ASYNC_LDS
    async_cp16(krow, &Ks[keyld][segld]);
    async_cp16(krow + 8, &Ks[keyld][segld + 8]);
#else
    {
      uint4 kv0 = *reinterpret_cast<const uint4*>(krow);
      uint4 kv1 = *reinterpret_cast<const uint4*>(krow + 8);
      *reinterpret_cast<uint4*>(&Ks[keyld][segld])     = kv0;
      *reinterpret_cast<uint4*>(&Ks[keyld][segld + 8]) = kv1;
    }
#endif
#pragma unroll
    for (int j = 0; j < 16; ++j) Vt[segld + j][keyld] = vv.b[j];  // transpose
#if HAVE_ASYNC_LDS
    wait_async0();  // K (and first-iteration Q) deposits done
#endif
    __syncthreads();

    // ---- scores: 16 q-rows x 32 keys, contraction over d=128 ----
    v8f sc[2];
    sc[0] = z8; sc[1] = z8;
#pragma unroll
    for (int ks = 0; ks < 4; ++ks) {
      v16bf aq = loadA(&Qs[wid * 16 + r][ks * 32], hi);
#pragma unroll
      for (int nt = 0; nt < 2; ++nt) {
        v16bf bk = loadB(&Ks[nt * 16 + r][ks * 32], hi);
        sc[nt] = wmma_bf16(aq, bk, sc[nt]);
      }
    }

    // ---- scale + causal mask + online softmax (DPP row reductions) ----
#pragma unroll
    for (int i = 0; i < 8; ++i) {
      const int sq = qb * 128 + wid * 16 + hi * 8 + i;
      float s0 = sc[0][i] * scale;
      float s1 = sc[1][i] * scale;
      if (kb * 32 + r > sq)      s0 = -1.0e9f;
      if (kb * 32 + 16 + r > sq) s1 = -1.0e9f;
      const float mx = red_max16(fmaxf(s0, s1));
      const float mnew = fmaxf(mrow[i], mx);
      const float p0 = __expf(s0 - mnew);
      const float p1 = __expf(s1 - mnew);
      const float sum = red_sum16(p0 + p1);
      const float corr = __expf(mrow[i] - mnew);
      lrow[i] = lrow[i] * corr + sum;
      mrow[i] = mnew;
#pragma unroll
      for (int dt = 0; dt < 8; ++dt) oacc[dt][i] = oacc[dt][i] * corr;
      Ps[wid][hi * 8 + i][r]      = f2bf(p0);  // C-layout -> [row][key]
      Ps[wid][hi * 8 + i][16 + r] = f2bf(p1);
    }
    asm volatile("s_wait_dscnt 0x0" ::: "memory");  // wave-private LDS RAW

    // ---- O += P(16x32) * V(32x128) ----
    v16bf ap = loadA(&Ps[wid][r][0], hi);
#pragma unroll
    for (int dt = 0; dt < 8; ++dt) {
      v16bf bv = loadB(&Vt[dt * 16 + r][0], hi);
      oacc[dt] = wmma_bf16(ap, bv, oacc[dt]);
    }
  }

  // ---- normalize and store as [S][HID] bf16 ----
#pragma unroll
  for (int i = 0; i < 8; ++i) {
    const int s = qb * 128 + wid * 16 + hi * 8 + i;
    const float inv = 1.0f / lrow[i];
#pragma unroll
    for (int dt = 0; dt < 8; ++dt) {
      Oa[(size_t)s * HID + h * DHEAD + dt * 16 + r] = f2bf(oacc[dt][i] * inv);
    }
  }
}

// ---------------------------------------------------------------------------
extern "C" void kernel_launch(void* const* d_in, const int* in_sizes, int n_in,
                              void* d_out, int out_size, void* d_ws, size_t ws_size,
                              hipStream_t stream) {
  (void)in_sizes; (void)n_in; (void)out_size; (void)ws_size;
  const float* hidden = (const float*)d_in[0];
  // d_in[1] = attention_mask (implicit causal), d_in[6] = position_ids (arange)
  const float* Wq = (const float*)d_in[2];
  const float* Wk = (const float*)d_in[3];
  const float* Wv = (const float*)d_in[4];
  const float* Wo = (const float*)d_in[5];
  float* out = (float*)d_out;

  char* ws = (char*)d_ws;
  const size_t SZ_ACT = (size_t)S_LEN * HID * 2;  // 16 MiB
  const size_t SZ_W   = (size_t)HID * HID * 2;    // 32 MiB
  bf16* hbf  = (bf16*)(ws);
  bf16* wqb  = (bf16*)(ws + SZ_ACT);
  bf16* wkb  = (bf16*)(ws + SZ_ACT + SZ_W);
  bf16* wvb  = (bf16*)(ws + SZ_ACT + 2 * SZ_W);
  bf16* wob  = (bf16*)(ws + SZ_ACT + 3 * SZ_W);
  bf16* qbuf = (bf16*)(ws + SZ_ACT + 4 * SZ_W);
  bf16* kbuf = (bf16*)(ws + 2 * SZ_ACT + 4 * SZ_W);
  bf16* vbuf = (bf16*)(ws + 3 * SZ_ACT + 4 * SZ_W);
  bf16* abuf = (bf16*)(ws + 4 * SZ_ACT + 4 * SZ_W);

  convert_rows<<<S_LEN, 256, 0, stream>>>(hidden, hbf, 0);
  convert_rows<<<HID, 256, 0, stream>>>(Wq, wqb, 1);  // fold permute_W
  convert_rows<<<HID, 256, 0, stream>>>(Wk, wkb, 1);  // fold permute_W
  convert_rows<<<HID, 256, 0, stream>>>(Wv, wvb, 0);
  convert_rows<<<HID, 256, 0, stream>>>(Wo, wob, 0);

  gemm128<false><<<dim3(32, 16, 3), 256, 0, stream>>>(
      hbf, wqb, wkb, wvb, qbuf, kbuf, vbuf, nullptr);
  flash_attn<<<dim3(16, 32), 256, 0, stream>>>(qbuf, kbuf, vbuf, abuf);
  gemm128<true><<<dim3(32, 16, 1), 256, 0, stream>>>(
      abuf, wob, nullptr, nullptr, nullptr, nullptr, nullptr, out);
}
